// MomentumDelta_3204045603517
// MI455X (gfx1250) — matrix operand order repaired and checked by reading, not compiled, not run on hardware
//
#include <hip/hip_runtime.h>

#define B_ 64
#define L_ 2048
#define H_ 256
#define H2_ 512
#define V_ 32000
#define CHUNK 64
#define NCHUNK 32
#define RBLK 4
#define BETA_F 0.9f
#define ONEMB 0.1f
#define EPSK 1e-6f
#define EPSLN 1e-5f

typedef _Float16 half_t;
typedef __attribute__((ext_vector_type(16))) _Float16 v16h;
typedef __attribute__((ext_vector_type(8))) float v8f;
typedef __attribute__((ext_vector_type(4))) unsigned int v4u;
typedef __attribute__((ext_vector_type(8))) int v8i_t;
typedef __attribute__((ext_vector_type(4))) int v4i_t;

static __device__ inline int laneid() { return threadIdx.x & 31; }

// ---- WMMA fragment helpers (CDNA5 16x16x32 f16 layouts) -------------------
// A fragment (16x32, MxK), source row-major [m][k], leading dim ld (elements)
static __device__ inline v16h ldA(const half_t* p, int ld, int m0, int k0) {
  int l = laneid();
  const half_t* r = p + (m0 + (l & 15)) * ld + k0 + 8 * (l >> 4);
  v16h a;
#pragma unroll
  for (int i = 0; i < 8; ++i) a[i] = r[i];
#pragma unroll
  for (int i = 0; i < 8; ++i) a[8 + i] = r[16 + i];
  return a;
}
// A fragment converted from f32 row-major source
static __device__ inline v16h ldA_f32(const float* p, int ld, int m0, int k0) {
  int l = laneid();
  const float* r = p + (m0 + (l & 15)) * ld + k0 + 8 * (l >> 4);
  v16h a;
#pragma unroll
  for (int i = 0; i < 8; ++i) a[i] = (half_t)r[i];
#pragma unroll
  for (int i = 0; i < 8; ++i) a[8 + i] = (half_t)r[16 + i];
  return a;
}
// B fragment (32x16, KxN): storage "n-major": src[n][k], leading dim ld
static __device__ inline v16h ldB(const half_t* p, int ld, int n0, int k0) {
  int l = laneid();
  const half_t* r = p + (n0 + (l & 15)) * ld + k0 + 8 * (l >> 4);
  v16h b;
#pragma unroll
  for (int i = 0; i < 8; ++i) b[i] = r[i];
#pragma unroll
  for (int i = 0; i < 8; ++i) b[8 + i] = r[16 + i];
  return b;
}
// C/D fragment <-> row-major f32 [m][n]
static __device__ inline v8f ldC(const float* p, int ld, int m0, int n0) {
  int l = laneid();
  const float* r = p + (m0 + 8 * (l >> 4)) * ld + n0 + (l & 15);
  v8f c;
#pragma unroll
  for (int v = 0; v < 8; ++v) c[v] = r[v * ld];
  return c;
}
static __device__ inline void stC(float* p, int ld, int m0, int n0, v8f c) {
  int l = laneid();
  float* r = p + (m0 + 8 * (l >> 4)) * ld + n0 + (l & 15);
#pragma unroll
  for (int v = 0; v < 8; ++v) r[v * ld] = c[v];
}

// ---- transpose + f32->f16 convert: in[R][C] -> out[C][R] ------------------
__global__ void k_transcvt(const float* in, half_t* out, int R, int C) {
  int idx = blockIdx.x * 256 + threadIdx.x;
  if (idx < R * C) {
    int r = idx / C, c = idx % C;
    out[c * R + r] = (half_t)in[r * C + c];
  }
}

// ---- encoder: gather -> FFN (WMMA) -> residual -> LayerNorm ---------------
__global__ __launch_bounds__(256) void k_encoder(
    const int* seq, const float* embed, const half_t* W1T, const float* b1,
    const half_t* W2T, const float* b2, const float* gamma, const float* beta_ln,
    half_t* hb, float* hlast) {
  __shared__ float  eF[16][260];
  __shared__ half_t eH[16][272];
  __shared__ half_t A1[16][528];
  __shared__ float  xF[16][264];
  __shared__ float  mv[16][2];

  const int tid = threadIdx.x;
  const int tb = blockIdx.x * 16;  // global token base (flattened [B*L])
  {
    int row = tid >> 4, part = tid & 15;
    int tg = tb + row;
    int id = seq[tg];
    const float* src = embed + (long)id * H_ + part * 16;
#pragma unroll
    for (int i = 0; i < 16; ++i) {
      float v = src[i];
      eF[row][part * 16 + i] = v;
      eH[row][part * 16 + i] = (half_t)v;
    }
  }
  __syncthreads();
  const int w = tid >> 5;
  // x@W1 + b1, relu  ->  A1 (f16)
  for (int q = 0; q < 4; ++q) {
    int n0 = (w * 4 + q) * 16;
    int col = n0 + (laneid() & 15);
    float bias = b1[col];
    v8f c;
#pragma unroll
    for (int v = 0; v < 8; ++v) c[v] = bias;
#pragma unroll
    for (int k = 0; k < 8; ++k) {
      v16h a = ldA(&eH[0][0], 272, 0, k * 32);
      v16h b = ldB(W1T, H_, n0, k * 32);
      c = __builtin_amdgcn_wmma_f32_16x16x32_f16(false, a, false, b, (short)0, c, false, false);
    }
    int mbase = 8 * (laneid() >> 4);
#pragma unroll
    for (int v = 0; v < 8; ++v) {
      float x = c[v] > 0.f ? c[v] : 0.f;
      A1[mbase + v][col] = (half_t)x;
    }
  }
  __syncthreads();
  // A1@W2 + b2 + residual -> xF
  for (int q = 0; q < 2; ++q) {
    int n0 = (w * 2 + q) * 16;
    int col = n0 + (laneid() & 15);
    float bias = b2[col];
    v8f c;
#pragma unroll
    for (int v = 0; v < 8; ++v) c[v] = bias;
#pragma unroll
    for (int k = 0; k < 16; ++k) {
      v16h a = ldA(&A1[0][0], 528, 0, k * 32);
      v16h b = ldB(W2T, H2_, n0, k * 32);
      c = __builtin_amdgcn_wmma_f32_16x16x32_f16(false, a, false, b, (short)0, c, false, false);
    }
    int mbase = 8 * (laneid() >> 4);
#pragma unroll
    for (int v = 0; v < 8; ++v)
      xF[mbase + v][col] = c[v] + eF[mbase + v][col];
  }
  __syncthreads();
  if (tid < 16) {
    float s = 0.f;
#pragma unroll 8
    for (int c = 0; c < H_; ++c) s += xF[tid][c];
    float mu = s * (1.0f / H_);
    float s2 = 0.f;
#pragma unroll 8
    for (int c = 0; c < H_; ++c) { float d = xF[tid][c] - mu; s2 += d * d; }
    mv[tid][0] = mu;
    mv[tid][1] = rsqrtf(s2 * (1.0f / H_) + EPSLN);
  }
  __syncthreads();
  {
    int row = tid >> 4, part = tid & 15;
    int tg = tb + row;
    float mu = mv[row][0], rs = mv[row][1];
    half_t* dst = hb + (long)tg * H_;
    bool last = ((tg & (L_ - 1)) == (L_ - 1));
    float* dl = hlast + (long)(tg >> 11) * H_;
#pragma unroll
    for (int i = 0; i < 16; ++i) {
      int cc = part * 16 + i;
      float h = (xF[row][cc] - mu) * rs * gamma[cc] + beta_ln[cc];
      dst[cc] = (half_t)h;
      if (last) dl[cc] = h;
    }
  }
}

// ---- chunked momentum delta-rule scan; M row-block persistent in LDS ------
__global__ __launch_bounds__(256) void k_scan(const half_t* hb, const float* hlast,
                                              float* y) {
  __shared__ float  Mblk[64][264];  // owned 64 rows of M (f32), persistent
  __shared__ half_t Kc[64][272];    // K chunk, row-major [token][h], 32B row pad
  __shared__ half_t Kt[256][72];    // K chunk transposed [h][token]
  __shared__ float  Uc[64][68];     // U = M K^T  [row][token]
  __shared__ float  Gc[64][68];     // Gram K K^T
  __shared__ float  Df[64][68];     // solved d vectors (f32), [row][token]
  __shared__ half_t Dh[64][72];     // weighted d (f16) for update matmul
  __shared__ float  nrm[64];
  __shared__ float  pw[CHUNK + 1];
  __shared__ float  hl[256];

  const int tid = threadIdx.x;
  const int w = tid >> 5;
  const int b = blockIdx.x >> 2;
  const int rb = blockIdx.x & 3;
  const int r0 = rb * 64;  // global row base of owned M block

  for (int i = tid; i < 64 * 264; i += 256) (&Mblk[0][0])[i] = 0.f;
  if (tid == 0) {
    float p = 1.f;
    for (int i = 0; i <= CHUNK; ++i) { pw[i] = p; p *= BETA_F; }
  }
  __syncthreads();

  for (int ch = 0; ch < NCHUNK; ++ch) {
    const int t0 = ch * CHUNK;
    const int ceff = ((L_ - 1) - t0 < CHUNK) ? (L_ - 1) - t0 : CHUNK;
#if __has_builtin(__builtin_amdgcn_tensor_load_to_lds)
    {  // Tensor Data Mover: DMA [ceff x 256] f16 tile into padded Kc rows.
      if (tid < 32) {  // one wave issues the descriptor
        unsigned long long ga =
            (unsigned long long)(hb + ((long)b * L_ + t0) * H_);
        unsigned lds = (unsigned)(unsigned long long)&Kc[0][0];
        v4u g0;
        g0[0] = 1u;  // count=1 valid user descriptor
        g0[1] = lds; // lds_addr
        g0[2] = (unsigned)(ga & 0xffffffffu);                        // global_addr[31:0]
        g0[3] = (unsigned)((ga >> 32) & 0x01ffffffu) | (2u << 30);   // ga[56:32] | type=2
        v8i_t g1;
        // data_size=2B (code 1), pad_enable, pad_interval=128 dwords (code 6),
        // pad_amount=8 dwords (code 7) -> reproduces the 272-half row stride.
        g1[0] = (int)((1u << 16) | (1u << 20) | (6u << 22) | (7u << 25));
        g1[1] = (int)(256u << 16);   // tensor_dim0 = 256 elements
        g1[2] = (int)(2048u << 16);  // tensor_dim1 = L rows
        g1[3] = (int)(256u << 16);   // tile_dim0 = 256 elements
        g1[4] = (int)(unsigned)ceff; // tile_dim1 = rows this chunk; tile_dim2=0
        g1[5] = (int)H_;             // tensor_dim0_stride = 256 elements
        g1[6] = 0;
        g1[7] = 0;
        v4i_t gz4;
        gz4[0] = gz4[1] = gz4[2] = gz4[3] = 0;
        v8i_t gz8;
#pragma unroll
        for (int i = 0; i < 8; ++i) gz8[i] = 0;
        // 6-arg form (clang-23 / therock-10.0 headers): extra int32x8 group
        // before the cache-policy immediate.
        __builtin_amdgcn_tensor_load_to_lds(g0, g1, gz4, gz4, gz8, 0);
      }
      // zero the (at most one) slot beyond the L-1 scanned tokens
      int row = tid >> 2, part = tid & 3;
      if (row >= ceff) {
        half_t* dst = &Kc[row][part * 64];
#pragma unroll
        for (int i = 0; i < 64; ++i) dst[i] = (half_t)0.f;
      }
      __builtin_amdgcn_s_wait_tensorcnt(0);
    }
#else
    {  // fallback: cooperative vector copy
      int row = tid >> 2, part = tid & 3;
      const half_t* src = hb + ((long)b * L_ + t0 + row) * H_ + part * 64;
      half_t* dst = &Kc[row][part * 64];
      if (row < ceff) {
        __builtin_prefetch(src + CHUNK * H_, 0, 1);
#pragma unroll
        for (int i = 0; i < 64; ++i) dst[i] = src[i];
      } else {
#pragma unroll
        for (int i = 0; i < 64; ++i) dst[i] = (half_t)0.f;
      }
    }
#endif
    __syncthreads();
    {  // transpose into Kt
      int h = tid;
#pragma unroll
      for (int i = 0; i < 64; ++i) Kt[h][i] = Kc[i][h];
    }
    // G = K K^T and U = M K^T : 2 tiles of each per wave
    for (int q = 0; q < 2; ++q) {
      int t = w * 2 + q;
      int mt = t >> 2, nt = t & 3;
      v8f g, u;
#pragma unroll
      for (int v = 0; v < 8; ++v) { g[v] = 0.f; u[v] = 0.f; }
#pragma unroll
      for (int k = 0; k < 8; ++k) {
        v16h bK = ldB(&Kc[0][0], 272, nt * 16, k * 32);
        v16h aK = ldA(&Kc[0][0], 272, mt * 16, k * 32);
        g = __builtin_amdgcn_wmma_f32_16x16x32_f16(false, aK, false, bK, (short)0, g, false, false);
        v16h aM = ldA_f32(&Mblk[0][0], 264, mt * 16, k * 32);
        u = __builtin_amdgcn_wmma_f32_16x16x32_f16(false, aM, false, bK, (short)0, u, false, false);
      }
      stC(&Gc[0][0], 68, mt * 16, nt * 16, g);
      stC(&Uc[0][0], 68, mt * 16, nt * 16, u);
    }
    __syncthreads();
    if (tid < CHUNK) nrm[tid] = 1.0f / (Gc[tid][tid] + EPSK);
    __syncthreads();
    // forward substitution, independent per owned h-component
    if (tid < 64) {
      int hg = r0 + tid;
      for (int i = 0; i < CHUNK; ++i) {
        float acc = 0.f;
        for (int j = 0; j < i; ++j) acc += pw[i - 1 - j] * Gc[j][i] * Df[tid][j];
        float kv = (float)Kc[i][hg];
        float di = kv - (pw[i] * Uc[tid][i] + ONEMB * acc) * nrm[i];
        Df[tid][i] = di;
        float wi = (i < ceff) ? ONEMB * pw[ceff - 1 - i] : 0.f;
        Dh[tid][i] = (half_t)(wi * di);
      }
    }
    __syncthreads();
    // M <- beta^ceff * M + Dh * K   (rank-64 update, 8 tiles per wave)
    float decay = pw[ceff];
    for (int q = 0; q < 8; ++q) {
      int t = w * 8 + q;
      int mt = t >> 4, nt = t & 15;
      v8f c = ldC(&Mblk[0][0], 264, mt * 16, nt * 16);
#pragma unroll
      for (int v = 0; v < 8; ++v) c[v] *= decay;
#pragma unroll
      for (int k = 0; k < 2; ++k) {
        v16h a = ldA(&Dh[0][0], 72, mt * 16, k * 32);
        v16h bb = ldB(&Kt[0][0], 72, nt * 16, k * 32);
        c = __builtin_amdgcn_wmma_f32_16x16x32_f16(false, a, false, bb, (short)0, c, false, false);
      }
      stC(&Mblk[0][0], 264, mt * 16, nt * 16, c);
    }
    __syncthreads();
  }
  // readout: y = M h_last (owned rows)
  hl[tid] = hlast[b * H_ + tid];
  __syncthreads();
  if (tid < 64) {
    float acc = 0.f;
#pragma unroll 8
    for (int c = 0; c < H_; ++c) acc += Mblk[tid][c] * hl[c];
    y[b * H_ + r0 + tid] = acc;
  }
}

// ---- z = y @ rp_W + rp_b (f16 output for next GEMM) -----------------------
__global__ __launch_bounds__(256) void k_z(const float* y, const half_t* rpWT,
                                           const float* rp_b, half_t* zH) {
  __shared__ half_t yH[64][272];
  int tid = threadIdx.x;
  for (int i = tid; i < 64 * H_; i += 256) {
    int r = i >> 8, c = i & 255;
    yH[r][c] = (half_t)y[i];
  }
  __syncthreads();
  int w = tid >> 5;
  for (int q = 0; q < 8; ++q) {
    int t = w * 8 + q;
    int mt = t >> 4, nt = t & 15;
    int n0 = nt * 16;
    int col = n0 + (laneid() & 15);
    float bias = rp_b[col];
    v8f c;
#pragma unroll
    for (int v = 0; v < 8; ++v) c[v] = bias;
#pragma unroll
    for (int k = 0; k < 8; ++k) {
      v16h a = ldA(&yH[0][0], 272, mt * 16, k * 32);
      v16h b = ldB(rpWT, H_, n0, k * 32);
      c = __builtin_amdgcn_wmma_f32_16x16x32_f16(false, a, false, b, (short)0, c, false, false);
    }
    int mb = mt * 16 + 8 * (laneid() >> 4);
#pragma unroll
    for (int v = 0; v < 8; ++v) zH[(mb + v) * H_ + col] = (half_t)c[v];
  }
}

// ---- out = z @ out_W + out_b ----------------------------------------------
__global__ __launch_bounds__(256) void k_out(const half_t* zH, const half_t* outWT,
                                             const float* out_b, float* out) {
  int tid = threadIdx.x;
  int w = tid >> 5;
  int nb = blockIdx.x * 64;
  for (int q = 0; q < 2; ++q) {
    int t = w * 2 + q;
    int mt = t >> 2, nt = t & 3;
    int n0 = nb + nt * 16;
    int col = n0 + (laneid() & 15);
    float bias = out_b[col];
    v8f c;
#pragma unroll
    for (int v = 0; v < 8; ++v) c[v] = bias;
#pragma unroll
    for (int k = 0; k < 8; ++k) {
      v16h a = ldA(zH, H_, mt * 16, k * 32);
      v16h b = ldB(outWT, H_, n0, k * 32);
      c = __builtin_amdgcn_wmma_f32_16x16x32_f16(false, a, false, b, (short)0, c, false, false);
    }
    int mb = mt * 16 + 8 * (laneid() >> 4);
#pragma unroll
    for (int v = 0; v < 8; ++v) out[(long)(mb + v) * V_ + col] = c[v];
  }
}

extern "C" void kernel_launch(void* const* d_in, const int* in_sizes, int n_in,
                              void* d_out, int out_size, void* d_ws, size_t ws_size,
                              hipStream_t stream) {
  const int*   seq    = (const int*)d_in[0];
  const float* embed  = (const float*)d_in[1];
  const float* W1     = (const float*)d_in[2];
  const float* b1     = (const float*)d_in[3];
  const float* W2     = (const float*)d_in[4];
  const float* b2     = (const float*)d_in[5];
  const float* gamma  = (const float*)d_in[6];
  const float* betaln = (const float*)d_in[7];
  const float* rpW    = (const float*)d_in[8];
  const float* rpb    = (const float*)d_in[9];
  const float* outW   = (const float*)d_in[10];
  const float* outb   = (const float*)d_in[11];
  float* out = (float*)d_out;

  char* ws = (char*)d_ws;
  half_t* hb    = (half_t*)(ws);                  // 67,108,864 B
  float*  hlast = (float*)(ws + 67108864);        //     65,536 B
  half_t* W1T   = (half_t*)(ws + 67174400);       //    262,144 B
  half_t* W2T   = (half_t*)(ws + 67436544);       //    262,144 B
  half_t* rpWT  = (half_t*)(ws + 67698688);       //    131,072 B
  half_t* outWT = (half_t*)(ws + 67829760);       // 16,384,000 B
  float*  yv    = (float*)(ws + 84213760);        //     65,536 B
  half_t* zH    = (half_t*)(ws + 84279296);       //     32,768 B

  k_transcvt<<<(256 * 512 + 255) / 256, 256, 0, stream>>>(W1, W1T, 256, 512);
  k_transcvt<<<(512 * 256 + 255) / 256, 256, 0, stream>>>(W2, W2T, 512, 256);
  k_transcvt<<<(256 * 256 + 255) / 256, 256, 0, stream>>>(rpW, rpWT, 256, 256);
  k_transcvt<<<(256 * 32000 + 255) / 256, 256, 0, stream>>>(outW, outWT, 256, 32000);
  k_encoder<<<(B_ * L_) / 16, 256, 0, stream>>>(seq, embed, W1T, b1, W2T, b2,
                                                gamma, betaln, hb, hlast);
  k_scan<<<B_ * RBLK, 256, 0, stream>>>(hb, hlast, yv);
  k_z<<<1, 256, 0, stream>>>(yv, rpWT, rpb, zH);
  k_out<<<V_ / 64, 256, 0, stream>>>(zH, outWT, outb, out);
}
